// AttentionLayer_89678917141128
// MI455X (gfx1250) — matrix operand order repaired and checked by reading, not compiled, not run on hardware
//
#include <hip/hip_runtime.h>
#include <hip/hip_bf16.h>

#define BATCH 8
#define TQ    128
#define TK    512
#define DFEAT 512
#define HID   128
#define VSZ   512

typedef __attribute__((ext_vector_type(16))) __bf16 v16bf;
typedef __attribute__((ext_vector_type(8)))  float  v8f;

// Load two consecutive f32 (8B load) and place as packed bf16 pair i of a fragment.
__device__ __forceinline__ void set_pair(v16bf& f, int i, const float* __restrict__ p) {
    float2 t = *(const float2*)p;
    f[2*i]   = (__bf16)t.x;
    f[2*i+1] = (__bf16)t.y;
}

// Build a full A/B-row fragment from 16 consecutive floats (contiguous k = 16h + s for both).
__device__ __forceinline__ v16bf frag_rowmajor_A(const float* __restrict__ p) {
    v16bf f;
    set_pair(f, 0, p +  0); set_pair(f, 1, p +  2);
    set_pair(f, 2, p +  4); set_pair(f, 3, p +  6);
    set_pair(f, 4, p +  8); set_pair(f, 5, p + 10);
    set_pair(f, 6, p + 12); set_pair(f, 7, p + 14);
    return f;
}
__device__ __forceinline__ v16bf frag_rowmajor_B(const float* __restrict__ p) {
    v16bf f;
    set_pair(f, 0, p +  0); set_pair(f, 1, p +  2);
    set_pair(f, 2, p +  4); set_pair(f, 3, p +  6);
    set_pair(f, 4, p +  8); set_pair(f, 5, p + 10);
    set_pair(f, 6, p + 12); set_pair(f, 7, p + 14);
    return f;
}

__device__ __forceinline__ float dev_tanh(float x) {
    float r;
    asm("v_tanh_f32 %0, %1" : "=v"(r) : "v"(x));
    return r;
}

// ---------------------------------------------------------------------------
// Kernel 1: q_proj = query @ Wq^T + bq ; k_proj = keys @ Wk^T + bk
// One 16x64 output strip per wave: 1 shared A fragment x 4 B fragments x
// 4 accumulators per K-chunk (A traffic amortized 4x).
// Wave-tiles: q: 64 row-tiles x 2 h-groups = 128 ; k: 256 x 2 = 512 ; total 640.
// ---------------------------------------------------------------------------
__global__ __launch_bounds__(256) void proj_kernel(
    const float* __restrict__ query, const float* __restrict__ keys,
    const float* __restrict__ Wq,    const float* __restrict__ bq,
    const float* __restrict__ Wk,    const float* __restrict__ bk,
    float* __restrict__ qp,          float* __restrict__ kp)
{
    const int wave = threadIdx.x >> 5;
    const int lane = threadIdx.x & 31;
    const int tile = blockIdx.x * 8 + wave;        // 0..639, grid sized exactly

    const float* in; const float* W; const float* bias; float* outp;
    int row0, h0;
    if (tile < 128) {
        row0 = (tile >> 1) * 16; h0 = (tile & 1) * 64;
        in = query; W = Wq; bias = bq; outp = qp;
    } else {
        int t = tile - 128;
        row0 = (t >> 1) * 16; h0 = (t & 1) * 64;
        in = keys; W = Wk; bias = bk; outp = kp;
    }

    const int n  = lane & 15;     // A: M index; B: N index; C/D: N index
    const int hl = lane >> 4;     // lane-half selector

    v8f acc0 = {}, acc1 = {}, acc2 = {}, acc3 = {};
    const float* arow = in + (size_t)(row0 + n) * DFEAT;        // A row (M = lane&15)
    const float* br0  = W  + (size_t)(h0 +  0 + n) * DFEAT;     // B col n of h-tile j
    const float* br1  = W  + (size_t)(h0 + 16 + n) * DFEAT;
    const float* br2  = W  + (size_t)(h0 + 32 + n) * DFEAT;
    const float* br3  = W  + (size_t)(h0 + 48 + n) * DFEAT;

    for (int kc = 0; kc < DFEAT; kc += 32) {
        __builtin_prefetch(arow + kc + 32 + 8 * hl, 0, 1);   // next K-chunk of A
        v16bf a = frag_rowmajor_A(arow + kc + 16 * hl);
        const int kb = kc + 16 * hl;
        v16bf b0 = frag_rowmajor_B(br0 + kb);
        v16bf b1 = frag_rowmajor_B(br1 + kb);
        v16bf b2 = frag_rowmajor_B(br2 + kb);
        v16bf b3 = frag_rowmajor_B(br3 + kb);
        acc0 = __builtin_amdgcn_wmma_f32_16x16x32_bf16(false, a, false, b0, (short)0, acc0, false, false);
        acc1 = __builtin_amdgcn_wmma_f32_16x16x32_bf16(false, a, false, b1, (short)0, acc1, false, false);
        acc2 = __builtin_amdgcn_wmma_f32_16x16x32_bf16(false, a, false, b2, (short)0, acc2, false, false);
        acc3 = __builtin_amdgcn_wmma_f32_16x16x32_bf16(false, a, false, b3, (short)0, acc3, false, false);
    }

    float* obase = outp + (size_t)(row0 + 8 * hl) * HID + h0 + n;
    {
        const float bv = bias[h0 + n];
        #pragma unroll
        for (int r = 0; r < 8; ++r) obase[(size_t)r * HID +  0] = acc0[r] + bv;
    }
    {
        const float bv = bias[h0 + 16 + n];
        #pragma unroll
        for (int r = 0; r < 8; ++r) obase[(size_t)r * HID + 16] = acc1[r] + bv;
    }
    {
        const float bv = bias[h0 + 32 + n];
        #pragma unroll
        for (int r = 0; r < 8; ++r) obase[(size_t)r * HID + 32] = acc2[r] + bv;
    }
    {
        const float bv = bias[h0 + 48 + n];
        #pragma unroll
        for (int r = 0; r < 8; ++r) obase[(size_t)r * HID + 48] = acc3[r] + bv;
    }
}

// ---------------------------------------------------------------------------
// Kernel 2: scores + softmax. One block per (b, q). 256 threads, 2 keys/thread.
// scores[k] = bo + sum_h Wo[h] * tanh(qp[b,q,h] + kp[b,k,h]); softmax over k.
// attn written directly into its slot of d_out (re-used as GEMM A in kernel 3).
// ---------------------------------------------------------------------------
__global__ __launch_bounds__(256) void score_softmax_kernel(
    const float* __restrict__ qp, const float* __restrict__ kp,
    const float* __restrict__ Wo, const float* __restrict__ bo,
    float* __restrict__ attn)
{
    __shared__ float s_q[HID];
    __shared__ float s_wo[HID];
    __shared__ float s_sc[TK];
    __shared__ float s_red[256];

    const int b = blockIdx.x >> 7;       // / TQ
    const int q = blockIdx.x & 127;      // % TQ
    const int t = threadIdx.x;

    if (t < HID) {
        s_q[t]  = qp[((size_t)b * TQ + q) * HID + t];
        s_wo[t] = Wo[t];
    }
    __syncthreads();

    const float bo0 = bo[0];
    #pragma unroll
    for (int kk = t; kk < TK; kk += 256) {
        const float* kr = kp + ((size_t)b * TK + kk) * HID;
        float s = 0.f;
        #pragma unroll 8
        for (int h = 0; h < HID; h += 4) {
            float4 kv = *(const float4*)(kr + h);
            s += s_wo[h + 0] * dev_tanh(s_q[h + 0] + kv.x);
            s += s_wo[h + 1] * dev_tanh(s_q[h + 1] + kv.y);
            s += s_wo[h + 2] * dev_tanh(s_q[h + 2] + kv.z);
            s += s_wo[h + 3] * dev_tanh(s_q[h + 3] + kv.w);
        }
        s_sc[kk] = s + bo0;
    }
    __syncthreads();

    // max reduction over 512 scores
    s_red[t] = fmaxf(s_sc[t], s_sc[t + 256]);
    __syncthreads();
    for (int off = 128; off > 0; off >>= 1) {
        if (t < off) s_red[t] = fmaxf(s_red[t], s_red[t + off]);
        __syncthreads();
    }
    const float mx = s_red[0];
    __syncthreads();

    // exp + sum reduction
    float e0 = __expf(s_sc[t] - mx);
    float e1 = __expf(s_sc[t + 256] - mx);
    s_sc[t] = e0; s_sc[t + 256] = e1;
    s_red[t] = e0 + e1;
    __syncthreads();
    for (int off = 128; off > 0; off >>= 1) {
        if (t < off) s_red[t] += s_red[t + off];
        __syncthreads();
    }
    const float inv = 1.f / s_red[0];

    float* arow = attn + ((size_t)b * TQ + q) * TK;
    arow[t]       = s_sc[t] * inv;
    arow[t + 256] = s_sc[t + 256] * inv;
}

// ---------------------------------------------------------------------------
// Kernel 3: context[b] = attn[b] (128x512) @ values[b] (512x512), bf16 WMMA.
// One 16x64 strip per wave (A fragment reused over 4 n-tiles).
// 8 batches x 8 m-tiles x 8 n-groups = 512 wave-tiles.
// ---------------------------------------------------------------------------
__global__ __launch_bounds__(256) void context_kernel(
    const float* __restrict__ attn, const float* __restrict__ values,
    float* __restrict__ ctx)
{
    const int wave = threadIdx.x >> 5;
    const int lane = threadIdx.x & 31;
    const int tile = blockIdx.x * 8 + wave;  // 0..511

    const int b  = tile >> 6;        // 64 wave-tiles per batch
    const int tt = tile & 63;
    const int m0 = (tt >> 3) * 16;   // 8 m-tiles
    const int n0 = (tt & 7) * 64;    // 8 n-groups of 64

    const int n  = lane & 15;
    const int hl = lane >> 4;

    v8f acc0 = {}, acc1 = {}, acc2 = {}, acc3 = {};
    const float* arow = attn + ((size_t)b * TQ + m0 + n) * TK;
    const float* vb   = values + (size_t)b * TK * VSZ + n0 + n;

    for (int kc = 0; kc < TK; kc += 32) {
        __builtin_prefetch(arow + kc + 32 + 8 * hl, 0, 1);
        v16bf a = frag_rowmajor_A(arow + kc + 16 * hl);

        // B = values[k][n0 + 16*j + n]; k-pairs strided by row (lane-coalesced)
        const float* pv = vb + (size_t)(kc + 16 * hl) * VSZ;
        v16bf b0, b1, b2, b3;
        #pragma unroll
        for (int v = 0; v < 16; ++v) {
            const float* pr = pv + (size_t)v * VSZ;
            b0[v] = (__bf16)pr[ 0];
            b1[v] = (__bf16)pr[16];
            b2[v] = (__bf16)pr[32];
            b3[v] = (__bf16)pr[48];
        }

        acc0 = __builtin_amdgcn_wmma_f32_16x16x32_bf16(false, a, false, b0, (short)0, acc0, false, false);
        acc1 = __builtin_amdgcn_wmma_f32_16x16x32_bf16(false, a, false, b1, (short)0, acc1, false, false);
        acc2 = __builtin_amdgcn_wmma_f32_16x16x32_bf16(false, a, false, b2, (short)0, acc2, false, false);
        acc3 = __builtin_amdgcn_wmma_f32_16x16x32_bf16(false, a, false, b3, (short)0, acc3, false, false);
    }

    float* obase = ctx + ((size_t)b * TQ + m0 + 8 * hl) * VSZ + n0 + n;
    #pragma unroll
    for (int r = 0; r < 8; ++r) {
        obase[(size_t)r * VSZ +  0] = acc0[r];
        obase[(size_t)r * VSZ + 16] = acc1[r];
        obase[(size_t)r * VSZ + 32] = acc2[r];
        obase[(size_t)r * VSZ + 48] = acc3[r];
    }
}

// ---------------------------------------------------------------------------
extern "C" void kernel_launch(void* const* d_in, const int* in_sizes, int n_in,
                              void* d_out, int out_size, void* d_ws, size_t ws_size,
                              hipStream_t stream) {
    const float* query  = (const float*)d_in[0];
    const float* keys   = (const float*)d_in[1];
    const float* values = (const float*)d_in[2];
    const float* Wq     = (const float*)d_in[3];
    const float* bq     = (const float*)d_in[4];
    const float* Wk     = (const float*)d_in[5];
    const float* bk     = (const float*)d_in[6];
    const float* Wo     = (const float*)d_in[7];
    const float* bo     = (const float*)d_in[8];

    float* ctx  = (float*)d_out;                               // 8*128*512
    float* attn = (float*)d_out + (size_t)BATCH * TQ * VSZ;    // 8*128*512

    float* qp = (float*)d_ws;                                  // 8*128*128
    float* kp = qp + (size_t)BATCH * TQ * HID;                 // 8*512*128

    proj_kernel<<<80, 256, 0, stream>>>(query, keys, Wq, bq, Wk, bk, qp, kp);
    score_softmax_kernel<<<BATCH * TQ, 256, 0, stream>>>(qp, kp, Wo, bo, attn);
    context_kernel<<<64, 256, 0, stream>>>(attn, values, ctx);
}
